// FRAPModel_81174881894475
// MI455X (gfx1250) — matrix operand order repaired and checked by reading, not compiled, not run on hardware
//
#include <hip/hip_runtime.h>
#include <hip/hip_bf16.h>

typedef __attribute__((ext_vector_type(16))) _Float16 v16h;
typedef __attribute__((ext_vector_type(8)))  float    v8f;

#define B_TOTAL 32768
#define WAVES   4          // waves per block; 4*16 = 64 batch rows per block

union Frag16 { v16h v; int4 q[2]; };
union PackH2 { _Float16 h[2]; unsigned u; };

// A-fragment K mapping for 16-bit 16x32 A tiles (ISA 7.12.2):
// lanes 0-15 hold K {0..7,16..23}; lanes 16-31 hold K {8..15,24..31} (+32*s)
__device__ __forceinline__ int kmapA(int s, int lane, int h) {
  int base = s * 32 + ((lane & 16) ? 8 : 0);
  return base + (h < 8 ? h : h + 8);
}

// Async global->LDS copy (CDNA5 GLOBAL_LOAD_ASYNC_TO_LDS_B128, ASYNCcnt).
// lds_off: LDS byte address (low 32 bits of generic pointer = LDS offset).
__device__ __forceinline__ void async_cp16(unsigned lds_off, const void* gsrc) {
  asm volatile("global_load_async_to_lds_b128 %0, %1, off"
               :: "v"(lds_off), "v"(gsrc) : "memory");
}
__device__ __forceinline__ void wait_async0() {
  asm volatile("s_wait_asynccnt 0x0" ::: "memory");
}

// ---------------------------------------------------------------------------
// Prep: pack weights into WMMA fragment order (f16), precompute h_s vectors,
// Hr table (tiny 16-row MLP), and bias/Wc pack.
// ---------------------------------------------------------------------------
__global__ __launch_bounds__(256, 1) void frap_prep(
    const float* __restrict__ Wv,  const float* __restrict__ bv,
    const float* __restrict__ Ws,  const float* __restrict__ bs,
    const float* __restrict__ Wm,  const float* __restrict__ rel,
    const float* __restrict__ Wd1, const float* __restrict__ Wd2,
    const float* __restrict__ Wr1, const float* __restrict__ br1,
    const float* __restrict__ Wr2, const float* __restrict__ br2,
    const float* __restrict__ bm,  const float* __restrict__ bd1,
    const float* __restrict__ bd2, const float* __restrict__ Wc,
    const float* __restrict__ bc,
    unsigned* __restrict__ wmF, unsigned* __restrict__ wd1F,
    unsigned* __restrict__ wd2F,
    _Float16* __restrict__ wvA, _Float16* __restrict__ bvA,
    _Float16* __restrict__ hsA, float* __restrict__ hrT,
    float* __restrict__ smallb)
{
  const int t = threadIdx.x;

  // Wm (128x64) B-fragments: [s=0..3][nt=0..3][lane][v] -> packed 2x f16
  for (int i = t; i < 4096; i += 256) {
    int v = i & 7, lane = (i >> 3) & 31, nt = (i >> 8) & 3, s = i >> 10;
    int K = s * 32 + ((lane & 16) ? 8 : 0) + (v < 4 ? 2 * v : 2 * v + 8);
    int n = nt * 16 + (lane & 15);
    PackH2 pk;
    pk.h[0] = (_Float16)Wm[K * 64 + n];
    pk.h[1] = (_Float16)Wm[(K + 1) * 64 + n];
    wmF[i] = pk.u;
  }
  // Wd1 / Wd2 (128x128) B-fragments: [s=0..3][nt=0..7][lane][v]
  for (int i = t; i < 8192; i += 256) {
    int v = i & 7, lane = (i >> 3) & 31, nt = (i >> 8) & 7, s = i >> 11;
    int K = s * 32 + ((lane & 16) ? 8 : 0) + (v < 4 ? 2 * v : 2 * v + 8);
    int n = nt * 16 + (lane & 15);
    PackH2 pk;
    pk.h[0] = (_Float16)Wd1[K * 128 + n];
    pk.h[1] = (_Float16)Wd1[(K + 1) * 128 + n];
    wd1F[i] = pk.u;
    pk.h[0] = (_Float16)Wd2[K * 128 + n];
    pk.h[1] = (_Float16)Wd2[(K + 1) * 128 + n];
    wd2F[i] = pk.u;
  }
  // Wv/bv in A-fragment order: [s=0..1][lane][h]
  for (int i = t; i < 1024; i += 256) {
    int h = i & 15, lane = (i >> 4) & 31, s = i >> 9;
    int K = kmapA(s, lane, h);
    wvA[i] = (_Float16)Wv[K];
    bvA[i] = (_Float16)bv[K];
  }
  // h_s candidates (s_mov is 0/1!): hs0=relu(bs), hs1=relu(Ws+bs), A-frag order
  for (int i = t; i < 2048; i += 256) {
    int h = i & 15, lane = (i >> 4) & 31, s = (i >> 9) & 1, sel = i >> 10;
    int K = kmapA(s, lane, h);
    float val = bs[K] + (sel ? Ws[K] : 0.f);
    hsA[i] = (_Float16)(val > 0.f ? val : 0.f);
  }
  // biases + Wc + bc packed: [0..63]=bm [64..191]=bd1 [192..319]=bd2 [320..447]=Wc [448]=bc
  for (int i = t; i < 512; i += 256) {
    float v = 0.f;
    if (i < 64)       v = bm[i];
    else if (i < 192) v = bd1[i - 64];
    else if (i < 320) v = bd2[i - 192];
    else if (i < 448) v = Wc[i - 320];
    else if (i == 448) v = bc[0];
    smallb[i] = v;
  }
  // Hr table: 16 rows through Wr1/Wr2 MLP
  __shared__ float h1[128];
  for (int pq = 0; pq < 16; ++pq) {
    __syncthreads();
    if (t < 128) {
      float a = br1[t];
      for (int k = 0; k < 32; ++k) a += rel[pq * 32 + k] * Wr1[k * 128 + t];
      h1[t] = a > 0.f ? a : 0.f;
    }
    __syncthreads();
    if (t < 128) {
      float a2 = br2[t];
      for (int k = 0; k < 128; ++k) a2 += h1[k] * Wr2[k * 128 + t];
      hrT[pq * 128 + t] = a2 > 0.f ? a2 : 0.f;
    }
  }
}

// ---------------------------------------------------------------------------
// Main fused kernel: 1 wave = 16 batch rows, all three GEMMs via WMMA f16.
// ---------------------------------------------------------------------------
__global__ __launch_bounds__(128, 1) void frap_main(
    const float* __restrict__ mc, const float* __restrict__ onehot,
    const unsigned* __restrict__ wmF, const unsigned* __restrict__ wd1F,
    const unsigned* __restrict__ wd2F,
    const _Float16* __restrict__ wvA, const _Float16* __restrict__ bvA,
    const _Float16* __restrict__ hsA, const float* __restrict__ hrT,
    const float* __restrict__ smallb, float* __restrict__ cflat)
{
  __shared__ alignas(16) unsigned s_wm [4 * 4 * 32 * 8];   // 16 KB
  __shared__ alignas(16) unsigned s_wd1[4 * 8 * 32 * 8];   // 32 KB
  __shared__ alignas(16) unsigned s_wd2[4 * 8 * 32 * 8];   // 32 KB
  __shared__ alignas(16) _Float16 s_wv [2 * 32 * 16];      // 2 KB
  __shared__ alignas(16) _Float16 s_bv [2 * 32 * 16];      // 2 KB
  __shared__ alignas(16) _Float16 s_hs [2 * 2 * 32 * 16];  // 4 KB
  __shared__ alignas(16) float    s_hr [16 * 128];         // 8 KB
  __shared__ alignas(16) float    s_small[512];            // 2 KB
  __shared__ alignas(16) float    s_mc [WAVES * 128];      // 2 KB  (mc tile)
  __shared__ alignas(16) int      s_ph [WAVES * 16];       // 256 B (phases)
  __shared__ alignas(16) _Float16 s_dp [WAVES * 64 * 72];  // 36 KB (padded rows)
  __shared__ alignas(16) _Float16 s_hd [WAVES * 16 * 136]; // 17 KB (padded rows)

  const int tid  = threadIdx.x;
  const int wave = tid >> 5;
  const int lane = tid & 31;
  const int bbase = (blockIdx.x * WAVES + wave) * 16;

  __builtin_prefetch(mc + (size_t)(blockIdx.x * WAVES * 16) * 8, 0, 3);
  __builtin_prefetch(onehot + (size_t)(blockIdx.x * WAVES * 16) * 4, 0, 3);

  // --- weight load: big fragment tables via CDNA5 async global->LDS DMA ----
  {
    auto acpy = [&](void* d, const void* s, int n16) {
      const char* ss = (const char*)s;
      const unsigned dl = (unsigned)(size_t)d;   // LDS byte offset
      for (int i = tid; i < n16; i += 128)
        async_cp16(dl + (unsigned)i * 16u, ss + (size_t)i * 16);
    };
    acpy(s_wm,  wmF,  1024);
    acpy(s_wd1, wd1F, 2048);
    acpy(s_wd2, wd2F, 2048);
    auto cpy = [&](void* d, const void* s, int n16) {
      int4* dd = (int4*)d; const int4* ss = (const int4*)s;
      for (int i = tid; i < n16; i += 128) dd[i] = ss[i];
    };
    cpy(s_wv,    wvA,    128);
    cpy(s_bv,    bvA,    128);
    cpy(s_hs,    hsA,    256);
    cpy(s_hr,    hrT,    512);
    cpy(s_small, smallb, 128);
  }
  // --- stage this wave's movement counts (128 f32) and 16 phases into LDS --
  {
    const float4 m4 = *(const float4*)(mc + (size_t)bbase * 8 + lane * 4);
    *(float4*)(s_mc + wave * 128 + lane * 4) = m4;
    const float4 oh4 = *(const float4*)(onehot + (size_t)(bbase + (lane & 15)) * 4);
    const int ph = (oh4.y > 0.5f ? 1 : 0) + (oh4.z > 0.5f ? 2 : 0) +
                   (oh4.w > 0.5f ? 3 : 0);
    if (lane < 16) s_ph[wave * 16 + lane] = ph;
  }
  wait_async0();
  __syncthreads();

  _Float16* dpW = s_dp + wave * 64 * 72;
  _Float16* hdW = s_hd + wave * 16 * 136;

  // -------- Phase 1: d_i = relu([h_v|h_s] @ Wm + bm); d_p = pairwise m-sum --
  for (int mt = 0; mt < 8; ++mt) {
    const int rowM = lane & 15;          // A-row index in tile
    const int m    = rowM & 7;
    const float mcv = s_mc[wave * 128 + mt * 16 + rowM];
    const int phase = s_ph[wave * 16 + (mt << 1) + (rowM >> 3)];
    const int sel = ((m >> 1) == phase) ? 1 : 0;

    Frag16 a[4];
    const _Float16 mch = (_Float16)mcv;
    for (int s = 0; s < 2; ++s) {        // h_v features, built in registers
      Frag16 w, bb;
      const int o = (s * 32 + lane) * 16;
      w.q[0]  = *(const int4*)(s_wv + o); w.q[1]  = *(const int4*)(s_wv + o + 8);
      bb.q[0] = *(const int4*)(s_bv + o); bb.q[1] = *(const int4*)(s_bv + o + 8);
      for (int i = 0; i < 16; ++i) {
        _Float16 x = (_Float16)(mch * w.v[i] + bb.v[i]);
        a[s].v[i] = (x > (_Float16)0.f) ? x : (_Float16)0.f;
      }
    }
    for (int s = 0; s < 2; ++s) {        // h_s features: table select
      const int o = (((sel << 1) | s) * 32 + lane) * 16;
      a[2 + s].q[0] = *(const int4*)(s_hs + o);
      a[2 + s].q[1] = *(const int4*)(s_hs + o + 8);
    }

    v8f acc[4] = {};
    for (int s = 0; s < 4; ++s)
      for (int nt = 0; nt < 4; ++nt) {
        Frag16 wfr;
        const unsigned* bp = s_wm + ((s * 4 + nt) * 32 + lane) * 8;
        wfr.q[0] = *(const int4*)bp; wfr.q[1] = *(const int4*)(bp + 4);
        acc[nt] = __builtin_amdgcn_wmma_f32_16x16x32_f16(
            false, a[s].v, false, wfr.v, (short)0, acc[nt], false, false);
      }

    // D-layout: vgpr v == m, lane half == which b; relu, pair-add -> d_p
    const int blD = (mt << 1) + (lane >> 4);
    for (int nt = 0; nt < 4; ++nt) {
      const int n = nt * 16 + (lane & 15);
      const float bmv = s_small[n];
      for (int p = 0; p < 4; ++p) {
        float x0 = acc[nt][2 * p]     + bmv; x0 = x0 > 0.f ? x0 : 0.f;
        float x1 = acc[nt][2 * p + 1] + bmv; x1 = x1 > 0.f ? x1 : 0.f;
        dpW[(blD * 4 + p) * 72 + n] = (_Float16)(x0 + x1);
      }
    }
  }
  __builtin_amdgcn_wave_barrier();

  // -------- Phase 2: per (p,q): Wd1 GEMM -> Wd2 GEMM -> *Hr -> dot Wc ------
  const float bcv = s_small[448];
  for (int pq = 0; pq < 16; ++pq) {
    const int p = pq >> 2, q = pq & 3;
    const int r = lane & 15;
    const int hk = (lane & 16) ? 8 : 0;

    Frag16 af[4];                        // A = [d_p[p] | d_p[q]], K=128
    for (int s = 0; s < 4; ++s) {
      const int pc = (s < 2) ? p : q;
      const int cbase = ((s & 1) << 5) + hk;
      const _Float16* sp = dpW + (r * 4 + pc) * 72 + cbase;
      af[s].q[0] = *(const int4*)sp;
      af[s].q[1] = *(const int4*)(sp + 16);
    }
    for (int nt = 0; nt < 8; ++nt) {     // Hd = relu(A @ Wd1 + bd1)
      v8f acc = {};
      for (int s = 0; s < 4; ++s) {
        Frag16 wfr;
        const unsigned* bp = s_wd1 + ((s * 8 + nt) * 32 + lane) * 8;
        wfr.q[0] = *(const int4*)bp; wfr.q[1] = *(const int4*)(bp + 4);
        acc = __builtin_amdgcn_wmma_f32_16x16x32_f16(
            false, af[s].v, false, wfr.v, (short)0, acc, false, false);
      }
      const int n = nt * 16 + (lane & 15);
      const float b1 = s_small[64 + n];
      const int rhalf = (lane & 16) ? 8 : 0;
      for (int v = 0; v < 8; ++v) {
        float x = acc[v] + b1; x = x > 0.f ? x : 0.f;
        hdW[(v + rhalf) * 136 + n] = (_Float16)x;
      }
    }
    __builtin_amdgcn_wave_barrier();

    Frag16 ag[4];                        // A = Hd, K=128
    for (int s = 0; s < 4; ++s) {
      const _Float16* sp = hdW + r * 136 + s * 32 + hk;
      ag[s].q[0] = *(const int4*)sp;
      ag[s].q[1] = *(const int4*)(sp + 16);
    }
    float csum[8] = {0.f, 0.f, 0.f, 0.f, 0.f, 0.f, 0.f, 0.f};
    for (int nt = 0; nt < 8; ++nt) {     // Hd2 = relu(Hd @ Wd2 + bd2); *Hr; .Wc
      v8f acc = {};
      for (int s = 0; s < 4; ++s) {
        Frag16 wfr;
        const unsigned* bp = s_wd2 + ((s * 8 + nt) * 32 + lane) * 8;
        wfr.q[0] = *(const int4*)bp; wfr.q[1] = *(const int4*)(bp + 4);
        acc = __builtin_amdgcn_wmma_f32_16x16x32_f16(
            false, ag[s].v, false, wfr.v, (short)0, acc, false, false);
      }
      const int n = nt * 16 + (lane & 15);
      const float b2 = s_small[192 + n];
      const float hr = s_hr[pq * 128 + n];
      const float wc = s_small[320 + n];
      for (int v = 0; v < 8; ++v) {
        float x = acc[v] + b2; x = x > 0.f ? x : 0.f;
        csum[v] += x * hr * wc;
      }
    }
    // reduce over the 16 lanes that cover the 128 columns (rows differ by half)
    for (int off = 1; off < 16; off <<= 1)
      for (int v = 0; v < 8; ++v)
        csum[v] += __shfl_xor(csum[v], off, 32);
    if ((lane & 15) == 0) {
      const int g = lane >> 4;
      for (int v = 0; v < 8; ++v) {
        float c = csum[v] + bcv; c = c > 0.f ? c : 0.f;
        cflat[(pq << 15) + bbase + v + (g << 3)] = c;  // p*4B + q*B + b
      }
    }
  }
}

// ---------------------------------------------------------------------------
// Gather: replicate reference's flat reshape: out[b',p'] = sum_q flat[16b'+4p'+q]
// ---------------------------------------------------------------------------
__global__ void frap_gather(const float* __restrict__ cflat,
                            float* __restrict__ out)
{
  const int t = blockIdx.x * blockDim.x + threadIdx.x;
  if (t < B_TOTAL * 4) {
    const float4 c = *(const float4*)(cflat + 4 * t);
    out[t] = c.x + c.y + c.z + c.w;
  }
}

extern "C" void kernel_launch(void* const* d_in, const int* in_sizes, int n_in,
                              void* d_out, int out_size, void* d_ws, size_t ws_size,
                              hipStream_t stream)
{
  const float* mc  = (const float*)d_in[0];
  const float* oh  = (const float*)d_in[1];
  const float* Wv  = (const float*)d_in[2];
  const float* bv  = (const float*)d_in[3];
  const float* Ws  = (const float*)d_in[4];
  const float* bs  = (const float*)d_in[5];
  const float* Wm  = (const float*)d_in[6];
  const float* bm  = (const float*)d_in[7];
  const float* rel = (const float*)d_in[8];
  const float* Wd1 = (const float*)d_in[9];
  const float* bd1 = (const float*)d_in[10];
  const float* Wd2 = (const float*)d_in[11];
  const float* bd2 = (const float*)d_in[12];
  const float* Wr1 = (const float*)d_in[13];
  const float* br1 = (const float*)d_in[14];
  const float* Wr2 = (const float*)d_in[15];
  const float* br2 = (const float*)d_in[16];
  const float* Wc  = (const float*)d_in[17];
  const float* bc  = (const float*)d_in[18];

  char* ws = (char*)d_ws;
  unsigned* wmF    = (unsigned*)(ws + 0);        // 16384 B
  unsigned* wd1F   = (unsigned*)(ws + 16384);    // 32768 B
  unsigned* wd2F   = (unsigned*)(ws + 49152);    // 32768 B
  _Float16* wvA    = (_Float16*)(ws + 81920);    // 2048 B
  _Float16* bvA    = (_Float16*)(ws + 83968);    // 2048 B
  _Float16* hsA    = (_Float16*)(ws + 86016);    // 4096 B
  float*    hrT    = (float*)   (ws + 90112);    // 8192 B
  float*    smallb = (float*)   (ws + 98304);    // 2048 B
  float*    cflat  = (float*)   (ws + 100352);   // 16*B*4 = 2 MB

  frap_prep<<<1, 256, 0, stream>>>(Wv, bv, Ws, bs, Wm, rel, Wd1, Wd2,
                                   Wr1, br1, Wr2, br2, bm, bd1, bd2, Wc, bc,
                                   wmF, wd1F, wd2F, wvA, bvA, hsA, hrT, smallb);
  frap_main<<<B_TOTAL / (WAVES * 16), 128, 0, stream>>>(
      mc, oh, wmF, wd1F, wd2F, wvA, bvA, hsA, hrT, smallb, cflat);
  frap_gather<<<(B_TOTAL * 4 + 255) / 256, 256, 0, stream>>>(
      cflat, (float*)d_out);
}
